// TrajDecoder_23733989278180
// MI455X (gfx1250) — compile-verified, matrix operand order checked by reading, not run on hardware
//
#include <hip/hip_runtime.h>
#include <hip/hip_bf16.h>

// ---------------------------------------------------------------------------
// Persistent WMMA-bf16 GRU decoder for MI455X (gfx1250, wave32).
//   B=16 batch rows == WMMA M dimension. H=2048 -> 128 column tiles of 16.
//   128 waves (16 blocks x 256 thr), each wave owns one 16-wide hidden tile
//   and its 3 gate tiles (r/z/n): 3 x 64 v_wmma_f32_16x16x32_bf16 per step.
//   w_hh as bf16 (24 MB) is L2-resident; h kept fp32 in registers (C layout),
//   bf16 shadow of h double-buffered in workspace for the A-matrix loads.
//   One atomic grid barrier per time step.
// ---------------------------------------------------------------------------

typedef __bf16 v16bf __attribute__((ext_vector_type(16)));
typedef __bf16 v8bf  __attribute__((ext_vector_type(8)));
typedef __bf16 v4bf  __attribute__((ext_vector_type(4)));
typedef float  v8f   __attribute__((ext_vector_type(8)));

constexpr int Bc    = 16;            // batch (== WMMA M)
constexpr int Hc    = 2048;          // hidden
constexpr int NT    = Hc / 16;       // 128 column tiles
constexpr int BLOCK = 256;           // 8 waves per block
constexpr int GRID  = NT / (BLOCK / 32);   // 16 blocks
constexpr int PAD   = 64;            // float stride between atomic slots (256B -> spread L2 channels)
constexpr int ACC_BUF = Bc * PAD;    // floats per accumulator buffer (triple buffered)

// workspace byte offsets
constexpr size_t OFF_BAR = 0;
constexpr size_t OFF_ACC = 256;
constexpr size_t OFF_HB  = 16384;            // 2 * 16*2048 bf16 = 128 KB
constexpr size_t OFF_WB  = 256 * 1024;       // 6144*2048 bf16   = 24 MB

// ---------------------------------------------------------------------------
__global__ void cvt_f32_bf16(const float* __restrict__ src,
                             unsigned short* __restrict__ dst, int n) {
  int i = (blockIdx.x * blockDim.x + threadIdx.x) * 4;
  if (i + 3 < n) {
    float4 v = *(const float4*)(src + i);
    v4bf o;
    o[0] = (__bf16)v.x; o[1] = (__bf16)v.y; o[2] = (__bf16)v.z; o[3] = (__bf16)v.w;
    *(v4bf*)(dst + i) = o;
  }
}

__global__ void init_state(const float* __restrict__ traj_z,
                           unsigned short* __restrict__ hb,
                           float* __restrict__ accum,
                           unsigned* __restrict__ bar) {
  int i = blockIdx.x * blockDim.x + threadIdx.x;
  if (i == 0) *bar = 0u;
  if (i < 3 * ACC_BUF) accum[i] = 0.0f;
  if (i < Bc * Hc) {
    __bf16 b = (__bf16)traj_z[i];
    hb[i] = __builtin_bit_cast(unsigned short, b);
  }
}

// ---------------------------------------------------------------------------
__device__ inline void grid_barrier(unsigned* bar, unsigned target) {
  __threadfence();                 // release
  __syncthreads();
  if (threadIdx.x == 0) {
    atomicAdd(bar, 1u);
    while (__hip_atomic_load(bar, __ATOMIC_RELAXED, __HIP_MEMORY_SCOPE_AGENT) < target) {
      __builtin_amdgcn_s_sleep(1);
    }
  }
  __syncthreads();
  __threadfence();                 // acquire (per-thread cache invalidate)
}

__device__ inline v16bf load_frag(const unsigned short* p) {
  // lane holds 8 contiguous K values, then 8 more at +16 (16-bit WMMA layout)
  v8bf lo = *(const v8bf*)p;
  v8bf hi = *(const v8bf*)(p + 16);
  return __builtin_shufflevector(lo, hi, 0, 1, 2, 3, 4, 5, 6, 7,
                                 8, 9, 10, 11, 12, 13, 14, 15);
}

__launch_bounds__(BLOCK, 1)
__global__ void traj_gru(const unsigned short* __restrict__ Wb,   // w_hh bf16 [3H][H]
                         unsigned short* __restrict__ hb,         // h bf16, 2 buffers [16][H]
                         float* __restrict__ accum,               // [3][ACC_BUF]
                         unsigned* __restrict__ bar,
                         const float* __restrict__ traj_z,
                         const float* __restrict__ traj_input,
                         const float* __restrict__ w_ih,
                         const float* __restrict__ b_ih,
                         const float* __restrict__ b_hh,
                         const float* __restrict__ w_traj,
                         const float* __restrict__ b_traj,
                         float* __restrict__ out,                 // [16][T]
                         int T) {
  const int lane = threadIdx.x & 31;
  const int warp = threadIdx.x >> 5;
  const int tile = blockIdx.x * (BLOCK / 32) + warp;   // 0..127
  const int n    = lane & 15;                          // N within tile / M for A loads
  const int hi   = lane >> 4;                          // half-wave select
  const int colb = tile * 16 + n;                      // hidden column owned by lane
  const int koff = 8 * hi;

  __shared__ float red[BLOCK / 32][16];

  // per-lane loop-invariant parameters
  const float wihr = w_ih[colb],        wihz = w_ih[Hc + colb],  wihn = w_ih[2 * Hc + colb];
  const float bihr = b_ih[colb],        bihz = b_ih[Hc + colb],  bihn = b_ih[2 * Hc + colb];
  const float bhhr = b_hh[colb],        bhhz = b_hh[Hc + colb],  bhhn = b_hh[2 * Hc + colb];
  const float wt   = w_traj[colb];
  const float btr  = b_traj[0];

  // fp32 master copy of this wave's h tile, in WMMA C layout:
  // element i <-> (row m = i + 8*hi, col = colb)
  float h[8], x[8];
#pragma unroll
  for (int i = 0; i < 8; ++i) h[i] = traj_z[(size_t)(i + 8 * hi) * Hc + colb];
#pragma unroll
  for (int i = 0; i < 8; ++i) x[i] = traj_input[(size_t)(i + 8 * hi) * T];  // x0

  const size_t rowR = (size_t)(0 * Hc + colb) * Hc + koff;
  const size_t rowZ = (size_t)(1 * Hc + colb) * Hc + koff;
  const size_t rowN = (size_t)(2 * Hc + colb) * Hc + koff;

  for (int t = 0; t < T; ++t) {
    // zero the accumulator buffer that will be written at step t+1
    // (triple buffering makes this race-free across the single barrier)
    if (blockIdx.x == 0 && threadIdx.x == 0) {
      float* zb = accum + ((t + 1) % 3) * ACC_BUF;
#pragma unroll
      for (int m = 0; m < 16; ++m) zb[m * PAD] = 0.0f;
    }

    const unsigned short* hrd = hb + (size_t)(t & 1) * (Bc * Hc);
    unsigned short*       hwr = hb + (size_t)((t + 1) & 1) * (Bc * Hc);
    const unsigned short* pa  = hrd + (size_t)n * Hc + koff;

    // gh = h @ w_hh.T for this wave's three 16x16 gate tiles
    v8f ar = {0.f, 0.f, 0.f, 0.f, 0.f, 0.f, 0.f, 0.f};
    v8f az = ar, an = ar;
#pragma unroll 4
    for (int k0 = 0; k0 < Hc; k0 += 32) {
      v16bf a  = load_frag(pa + k0);
      v16bf br = load_frag(Wb + rowR + k0);
      v16bf bz = load_frag(Wb + rowZ + k0);
      v16bf bn = load_frag(Wb + rowN + k0);
      ar = __builtin_amdgcn_wmma_f32_16x16x32_bf16(false, a, false, br, (short)0, ar, false, false);
      az = __builtin_amdgcn_wmma_f32_16x16x32_bf16(false, a, false, bz, (short)0, az, false, false);
      an = __builtin_amdgcn_wmma_f32_16x16x32_bf16(false, a, false, bn, (short)0, an, false, false);
    }

    // gates + state update (fp32)
    float dotp[8];
#pragma unroll
    for (int i = 0; i < 8; ++i) {
      float xm  = x[i];
      float gr  = ar[i] + bhhr + xm * wihr + bihr;
      float gz  = az[i] + bhhz + xm * wihz + bihz;
      float hn  = an[i] + bhhn;                 // gh_n includes b_hh
      float gin = xm * wihn + bihn;             // gi_n
      float r   = __builtin_amdgcn_rcpf(1.0f + __expf(-gr));
      float z   = __builtin_amdgcn_rcpf(1.0f + __expf(-gz));
      float nn  = tanhf(gin + r * hn);
      float hv  = (1.0f - z) * nn + z * h[i];
      h[i] = hv;
      __bf16 hb16 = (__bf16)hv;
      hwr[(size_t)(i + 8 * hi) * Hc + colb] = __builtin_bit_cast(unsigned short, hb16);
      dotp[i] = hv * wt;                        // partial of h_new @ w_traj.T
    }

    // reduce dot partial over the 16 lanes of this half-wave (N dimension)
#pragma unroll
    for (int i = 0; i < 8; ++i) {
      float v = dotp[i];
      v += __shfl_xor(v, 8, 32);
      v += __shfl_xor(v, 4, 32);
      v += __shfl_xor(v, 2, 32);
      v += __shfl_xor(v, 1, 32);
      dotp[i] = v;
    }
    if (n == 0) {
#pragma unroll
      for (int i = 0; i < 8; ++i) red[warp][8 * hi + i] = dotp[i];
    }
    __syncthreads();
    if (threadIdx.x < 16) {
      float s = 0.0f;
#pragma unroll
      for (int w = 0; w < BLOCK / 32; ++w) s += red[w][threadIdx.x];
      atomicAdd(accum + (t % 3) * ACC_BUF + threadIdx.x * PAD, s);
    }

    // device-wide step boundary
    grid_barrier(bar, (unsigned)((t + 1) * gridDim.x));

    // every wave redundantly advances x (needed for next step's gi)
    const float* ac = accum + (t % 3) * ACC_BUF;
#pragma unroll
    for (int i = 0; i < 8; ++i) x[i] += ac[(i + 8 * hi) * PAD] + btr;

    // one wave emits y_t = x_{t+1}
    if (blockIdx.x == 0 && warp == 0 && n == 0) {
#pragma unroll
      for (int i = 0; i < 8; ++i) out[(size_t)(i + 8 * hi) * T + t] = x[i];
    }
  }
}

// ---------------------------------------------------------------------------
extern "C" void kernel_launch(void* const* d_in, const int* in_sizes, int n_in,
                              void* d_out, int out_size, void* d_ws, size_t ws_size,
                              hipStream_t stream) {
  const float* traj_z     = (const float*)d_in[0];
  const float* traj_input = (const float*)d_in[1];
  const float* w_ih       = (const float*)d_in[2];
  const float* b_ih       = (const float*)d_in[3];
  const float* w_hh       = (const float*)d_in[4];
  const float* b_hh       = (const float*)d_in[5];
  const float* w_traj     = (const float*)d_in[6];
  const float* b_traj     = (const float*)d_in[7];
  float* out = (float*)d_out;
  const int T = in_sizes[1] / Bc;   // 512

  char* ws = (char*)d_ws;
  unsigned*       bar   = (unsigned*)(ws + OFF_BAR);
  float*          accum = (float*)(ws + OFF_ACC);
  unsigned short* hb    = (unsigned short*)(ws + OFF_HB);
  unsigned short* Wb    = (unsigned short*)(ws + OFF_WB);

  // one-time (per launch) weight conversion: 3H x H fp32 -> bf16 (L2-resident)
  const int nW = 3 * Hc * Hc;
  cvt_f32_bf16<<<nW / (256 * 4), 256, 0, stream>>>(w_hh, Wb, nW);
  // init barrier counter, dot accumulators, bf16 copy of h0
  init_state<<<(Bc * Hc + 255) / 256, 256, 0, stream>>>(traj_z, hb, accum, bar);
  // persistent recurrent kernel: 128 waves, 1 grid barrier per step
  traj_gru<<<GRID, BLOCK, 0, stream>>>(Wb, hb, accum, bar, traj_z, traj_input,
                                       w_ih, b_ih, b_hh, w_traj, b_traj, out, T);
}
